// VectorQuantizer_66348654788807
// MI455X (gfx1250) — matrix operand order repaired and checked by reading, not compiled, not run on hardware
//
#include <hip/hip_runtime.h>
#include <hip/hip_bf16.h>

typedef __attribute__((ext_vector_type(16))) __bf16 v16bf;
typedef __attribute__((ext_vector_type(8)))  float  v8f;

#define VQ_D 64
#define VQ_K 128
#define ROWS_TOTAL (256 * 64 * 64)      // 1,048,576 rows of 64
#define WAVES_PER_BLOCK 8
#define TILES_PER_WAVE 4
#define VQ_BLOCKS (ROWS_TOTAL / (16 * WAVES_PER_BLOCK * TILES_PER_WAVE))   // 2048
#define NUMEL ((float)(ROWS_TOTAL) * (float)VQ_D)

// Main kernel: distances via bf16 WMMA, argmin, gather codes, partial loss.
__global__ __launch_bounds__(256) void vq_main_kernel(
    const float* __restrict__ X,     // [ROWS_TOTAL, 64]
    const float* __restrict__ E,     // [64, 128] row-major E[d*128 + k]
    float* __restrict__ Out,         // [ROWS_TOTAL, 64]
    float* __restrict__ partials)    // [VQ_BLOCKS]
{
    // LDS staging: f32 transposed embeddings, swizzled bf16 B fragments, code norms.
    __shared__ __align__(16) float  sEt[VQ_K * VQ_D];          // 32 KB  E_t[k][d]
    __shared__ __align__(32) __bf16 sB[2 * 8 * 32 * 16];       // 16 KB  [kt][ct][lane][16]
    __shared__ float sEbn[VQ_K];                               // 512 B
    __shared__ float sPartial[WAVES_PER_BLOCK];

    // ---- Stage embeddings (all 256 threads) ----
    for (int idx = threadIdx.x; idx < VQ_D * VQ_K; idx += blockDim.x) {
        int d = idx >> 7;            // row of E (dim)
        int k = idx & 127;           // code index
        float v = E[idx];
        sEt[k * VQ_D + d] = v;
        // bf16 B-matrix swizzle: column k of the 32x16 (KxN) B tile lives in
        // lane (k&15) [+16 for K 8..15,24..31 within the k-step], elems 0..7 = K 8h..8h+7, 8..15 = K 16+8h..16+8h+7.
        int kt   = d >> 5;
        int dd   = d & 31;
        int lane = (k & 15) + ((dd & 8) ? 16 : 0);
        int elem = (dd & 7) | ((dd & 16) >> 1);
        int ct   = k >> 4;
        sB[(((kt * 8 + ct) * 32 + lane) << 4) + elem] = (__bf16)v;
    }
    __syncthreads();
    for (int k = threadIdx.x; k < VQ_K; k += blockDim.x) {
        float s = 0.f;
        #pragma unroll 8
        for (int d = 0; d < VQ_D; ++d) { float e = sEt[k * VQ_D + d]; s += e * e; }
        sEbn[k] = s;
    }
    __syncthreads();

    const int wave = threadIdx.x >> 5;
    const int lane = threadIdx.x & 31;
    const int lr   = lane & 15;      // lane within half (N index / row index)
    const int h    = lane >> 4;      // half select
    float lossAcc = 0.f;

    for (int t = 0; t < TILES_PER_WAVE; ++t) {
        const long tile = ((long)blockIdx.x * WAVES_PER_BLOCK + wave) * TILES_PER_WAVE + t;
        const long r0   = tile * 16;

        if (t + 1 < TILES_PER_WAVE)
            __builtin_prefetch(X + (r0 + 16 + lr) * VQ_D, 0, 1);   // global_prefetch_b8

        // ---- A fragments: row r0+lr, per-half d slices, f32 -> bf16 ----
        const float* Xrow = X + (r0 + lr) * VQ_D;
        v16bf a[2];
        #pragma unroll
        for (int kt = 0; kt < 2; ++kt) {
            const int dbase = kt * 32 + h * 8;
            const float4 f0 = ((const float4*)(Xrow + dbase))[0];       // elems 0..3
            const float4 f1 = ((const float4*)(Xrow + dbase))[1];       // elems 4..7
            const float4 f2 = ((const float4*)(Xrow + dbase + 16))[0];  // elems 8..11
            const float4 f3 = ((const float4*)(Xrow + dbase + 16))[1];  // elems 12..15
            a[kt][0]=(__bf16)f0.x;  a[kt][1]=(__bf16)f0.y;  a[kt][2]=(__bf16)f0.z;  a[kt][3]=(__bf16)f0.w;
            a[kt][4]=(__bf16)f1.x;  a[kt][5]=(__bf16)f1.y;  a[kt][6]=(__bf16)f1.z;  a[kt][7]=(__bf16)f1.w;
            a[kt][8]=(__bf16)f2.x;  a[kt][9]=(__bf16)f2.y;  a[kt][10]=(__bf16)f2.z; a[kt][11]=(__bf16)f2.w;
            a[kt][12]=(__bf16)f3.x; a[kt][13]=(__bf16)f3.y; a[kt][14]=(__bf16)f3.z; a[kt][15]=(__bf16)f3.w;
        }

        // ---- G tile = X_tile @ E : 8 column tiles x 2 k-steps of WMMA bf16 ----
        v8f acc[8];
        #pragma unroll
        for (int ct = 0; ct < 8; ++ct) {
            v8f c = {};
            v16bf b0 = *(const v16bf*)(sB + (((0 * 8 + ct) * 32 + lane) << 4));
            c = __builtin_amdgcn_wmma_f32_16x16x32_bf16(false, a[0], false, b0,
                                                        (short)0, c, false, false);
            v16bf b1 = *(const v16bf*)(sB + (((1 * 8 + ct) * 32 + lane) << 4));
            c = __builtin_amdgcn_wmma_f32_16x16x32_bf16(false, a[1], false, b1,
                                                        (short)0, c, false, false);
            acc[ct] = c;
        }

        // ---- Per-row argmin of (||e||^2 - 2 x.e), gather code, loss ----
        #pragma unroll
        for (int j = 0; j < 8; ++j) {
            float bv = 3.4e38f; int bi = 0;
            #pragma unroll
            for (int ct = 0; ct < 8; ++ct) {
                const int col = ct * 16 + lr;            // C/D layout: N = lane%16
                const float s = sEbn[col] - 2.0f * acc[ct][j];
                if (s < bv) { bv = s; bi = col; }        // cols ascend -> first-min kept
            }
            #pragma unroll
            for (int m = 1; m < 16; m <<= 1) {           // reduce within the 16-lane half
                const float ov = __shfl_xor(bv, m, 32);
                const int   oi = __shfl_xor(bi, m, 32);
                if (ov < bv || (ov == bv && oi < bi)) { bv = ov; bi = oi; }
            }
            // row m = j (lanes 0..15) or j+8 (lanes 16..31) per C/D layout
            const long row = r0 + j + 8 * h;
            const float4 q = ((const float4*)(sEt + bi * VQ_D))[lr];
            const float4 x = ((const float4*)(X + row * VQ_D))[lr];
            ((float4*)(Out + row * VQ_D))[lr] = q;       // 16 lanes -> 256B contiguous
            const float d0 = q.x - x.x, d1 = q.y - x.y, d2 = q.z - x.z, d3 = q.w - x.w;
            lossAcc += d0 * d0 + d1 * d1 + d2 * d2 + d3 * d3;
        }
    }

    // ---- Deterministic loss reduction: wave shuffle -> LDS -> block partial ----
    #pragma unroll
    for (int m = 1; m < 32; m <<= 1) lossAcc += __shfl_xor(lossAcc, m, 32);
    if (lane == 0) sPartial[wave] = lossAcc;
    __syncthreads();
    if (threadIdx.x == 0) {
        float s = 0.f;
        #pragma unroll
        for (int w = 0; w < WAVES_PER_BLOCK; ++w) s += sPartial[w];
        partials[blockIdx.x] = s;
    }
}

// Final reduction: fixed-order sum of block partials -> scalar loss.
__global__ __launch_bounds__(256) void vq_loss_reduce_kernel(
    const float* __restrict__ partials, int n, float* __restrict__ lossOut)
{
    __shared__ float sm[256];
    float s = 0.f;
    for (int i = threadIdx.x; i < n; i += 256) s += partials[i];
    sm[threadIdx.x] = s;
    __syncthreads();
    for (int st = 128; st > 0; st >>= 1) {
        if (threadIdx.x < st) sm[threadIdx.x] += sm[threadIdx.x + st];
        __syncthreads();
    }
    if (threadIdx.x == 0)
        lossOut[0] = 1.25f * sm[0] / NUMEL;   // q_loss + 0.25*e_loss, both == SSD/numel
}

extern "C" void kernel_launch(void* const* d_in, const int* in_sizes, int n_in,
                              void* d_out, int out_size, void* d_ws, size_t ws_size,
                              hipStream_t stream) {
    const float* X = (const float*)d_in[0];   // [256,64,64,64] f32
    const float* E = (const float*)d_in[1];   // [64,128] f32
    float* Out      = (float*)d_out;          // quantized, then 1 loss float
    float* loss     = Out + (out_size - 1);   // 67,108,864 quantized elems + loss
    float* partials = (float*)d_ws;           // VQ_BLOCKS floats (8 KB)

    vq_main_kernel<<<VQ_BLOCKS, 256, 0, stream>>>(X, E, Out, partials);
    vq_loss_reduce_kernel<<<1, 256, 0, stream>>>(partials, VQ_BLOCKS, loss);
}
